// MaskROI_841813590620
// MI455X (gfx1250) — compile-verified, matrix operand order checked by reading, not run on hardware
//
#include <hip/hip_runtime.h>

#ifndef __has_builtin
#define __has_builtin(x) 0
#endif

#define NBOX 1500
#define NFG 80
#define NTOT (NBOX * NFG)      // 120000
#define WRDS 48                // 48*32 = 1536 >= 1500 bits per mask row
#define KEYPAD 8192            // zero padding so k_select needs no bounds checks
#define SCORE_TH 0.05f
#define NMS_TH 0.5f
#define MAXDET 100

typedef int v8i __attribute__((ext_vector_type(8)));
typedef int v4i __attribute__((ext_vector_type(4)));

// ---------- gfx1250 async load-to-LDS (guarded; falls back to plain copy) ----------
#if __has_builtin(__builtin_amdgcn_global_load_async_to_lds_b128)
#define ASYNC_LDS 1
typedef __attribute__((address_space(1))) v4i g_v4i;   // global 128-bit vector
typedef __attribute__((address_space(3))) v4i l_v4i;   // LDS 128-bit vector
__device__ __forceinline__ l_v4i* as_lds(void* p) {
  // low 32 bits of a generic LDS address are the LDS byte offset
  return (l_v4i*)(unsigned)(unsigned long long)p;
}
__device__ __forceinline__ g_v4i* as_glb(const void* p) {
  return (g_v4i*)(unsigned long long)p;
}
#else
#define ASYNC_LDS 0
#endif

__device__ __forceinline__ void wait_async_lds() {
#if __has_builtin(__builtin_amdgcn_s_wait_asynccnt)
  __builtin_amdgcn_s_wait_asynccnt(0);
#else
  asm volatile("s_wait_asynccnt 0" ::: "memory");
#endif
}

// ---------------- Kernel 0: zero header + key buffer (harness never re-zeroes ws) ----------------
__global__ void k_init(unsigned* __restrict__ hdr, unsigned* __restrict__ keys) {
  int idx = blockIdx.x * 256 + threadIdx.x;
  if (idx == 0) { hdr[0] = 0u; hdr[1] = 0u; }
  if (idx < NTOT + KEYPAD) keys[idx] = 0u;
}

// ---------------- Kernel 1: box decode + clip, gather fg scores ----------------
__global__ void k_decode(const float* __restrict__ rois, const float* __restrict__ deltas,
                         const float* __restrict__ cls_prob, const float* __restrict__ im_info,
                         float* __restrict__ decB, float* __restrict__ scores) {
  int idx = blockIdx.x * 256 + threadIdx.x;
  if (idx >= NTOT) return;
  int c = idx / NBOX;     // fg class index 0..79 -> class c+1
  int i = idx % NBOX;
  float x1 = rois[i * 5 + 1], y1 = rois[i * 5 + 2];
  float x2 = rois[i * 5 + 3], y2 = rois[i * 5 + 4];
  float w = x2 - x1 + 1.0f, h = y2 - y1 + 1.0f;
  float cx = x1 + 0.5f * w, cy = y1 + 0.5f * h;
  int cls = c + 1;
  const float* d = deltas + (size_t)i * (4 * 81) + (size_t)cls * 4;
  float dx = d[0] / 10.0f, dy = d[1] / 10.0f, dw = d[2] / 5.0f, dh = d[3] / 5.0f;
  float pcx = cx + w * dx, pcy = cy + h * dy;
  float pw = w * expf(dw), ph = h * expf(dh);
  float H = im_info[0], W = im_info[1];
  float lx = W - 1.0f, ly = H - 1.0f;
  float bx1 = fminf(fmaxf(pcx - 0.5f * pw, 0.0f), lx);
  float by1 = fminf(fmaxf(pcy - 0.5f * ph, 0.0f), ly);
  float bx2 = fminf(fmaxf(pcx + 0.5f * pw - 1.0f, 0.0f), lx);
  float by2 = fminf(fmaxf(pcy + 0.5f * ph - 1.0f, 0.0f), ly);
  float* o = decB + (size_t)idx * 4;
  o[0] = bx1; o[1] = by1; o[2] = bx2; o[3] = by2;
  scores[idx] = cls_prob[(size_t)i * 81 + cls];
}

// ---------------- Kernel 2: stable descending rank + scatter into sorted order ----------------
__global__ void k_rank_scatter(const float* __restrict__ decB, const float* __restrict__ scores,
                               float* __restrict__ sB, float* __restrict__ sS) {
  int idx = blockIdx.x * 256 + threadIdx.x;
  if (idx >= NTOT) return;
  int c = idx / NBOX, i = idx % NBOX;
  const float* sc = scores + (size_t)c * NBOX;
  float si = sc[i];
  int rank = 0;
  for (int j = 0; j < NBOX; ++j) {
    float sj = sc[j];
    rank += (int)((sj > si) || ((sj == si) && (j < i)));   // stable argsort(-s)
  }
  int dst = c * NBOX + rank;
  const float4 b = ((const float4*)decB)[idx];
  ((float4*)sB)[dst] = b;
  sS[dst] = si;
}

// ---------------- Kernel 3: pairwise IoU suppression bit-mask (LDS staged) ----------------
__global__ __launch_bounds__(256) void k_iou_mask(const float* __restrict__ sB,
                                                  unsigned* __restrict__ M) {
  __shared__ float4 bsm[NBOX];
  __shared__ float areaSm[NBOX];
  int c = blockIdx.x / 12;
  int tile = blockIdx.x % 12;
  int i0 = tile * 128;
  int rows = (NBOX - i0 < 128) ? (NBOX - i0) : 128;
  int tid = threadIdx.x;
  const float4* src = (const float4*)(sB + (size_t)c * NBOX * 4);
  for (int i = tid; i < NBOX; i += 256) {
#if ASYNC_LDS
    __builtin_amdgcn_global_load_async_to_lds_b128(as_glb(src + i), as_lds((void*)&bsm[i]), 0, 0);
#else
    bsm[i] = src[i];
#endif
  }
#if ASYNC_LDS
  wait_async_lds();
#endif
  __syncthreads();
  for (int i = tid; i < NBOX; i += 256) {
    float4 b = bsm[i];
    areaSm[i] = (b.z - b.x + 1.0f) * (b.w - b.y + 1.0f);
  }
  __syncthreads();
  int total = rows * WRDS;
  for (int p = tid; p < total; p += 256) {
    int ri = p / WRDS, w = p % WRDS;
    int i = i0 + ri;
    float4 bi = bsm[i];
    float ai = areaSm[i];
    unsigned bits = 0u;
    int jb = w * 32;
    if (jb + 32 <= NBOX) {                 // 46 of 48 words: no bounds checks
      #pragma unroll
      for (int k = 0; k < 32; ++k) {
        int j = jb + k;
        float4 bj = bsm[j];
        float iw = fminf(bi.z, bj.z) - fmaxf(bi.x, bj.x) + 1.0f;
        float ih = fminf(bi.w, bj.w) - fmaxf(bi.y, bj.y) + 1.0f;
        iw = fmaxf(iw, 0.0f);
        ih = fmaxf(ih, 0.0f);
        float inter = iw * ih;
        float iou = inter / (ai + areaSm[j] - inter);
        if (iou > NMS_TH) bits |= (1u << k);
      }
    } else {                               // boundary word
      for (int k = 0; k < 32; ++k) {
        int j = jb + k;
        if (j < NBOX) {
          float4 bj = bsm[j];
          float iw = fminf(bi.z, bj.z) - fmaxf(bi.x, bj.x) + 1.0f;
          float ih = fminf(bi.w, bj.w) - fmaxf(bi.y, bj.y) + 1.0f;
          iw = fmaxf(iw, 0.0f);
          ih = fmaxf(ih, 0.0f);
          float inter = iw * ih;
          float iou = inter / (ai + areaSm[j] - inter);
          if (iou > NMS_TH) bits |= (1u << k);
        }
      }
    }
    M[((size_t)c * NBOX + i) * WRDS + w] = bits;
  }
}

// ---------------- Kernel 4: sequential NMS scan, one wave32 per class ----------------
__global__ void k_nms_scan(const float* __restrict__ sS, const unsigned* __restrict__ M,
                           unsigned* __restrict__ keepw) {
  int c = blockIdx.x;
  int lane = threadIdx.x;   // 32 threads, 1 wave
  const float* sc = sS + (size_t)c * NBOX;
  unsigned v0 = 0u, v1 = 0u;        // valid bits: words lane, lane+32
  for (int k = 0; k < 32; ++k) {
    int ia = lane * 32 + k;
    if (ia < NBOX && sc[ia] > SCORE_TH) v0 |= (1u << k);
    int ib = (lane + 32) * 32 + k;
    if (ib < NBOX && sc[ib] > SCORE_TH) v1 |= (1u << k);
  }
  unsigned sup0 = 0u, sup1 = 0u, kp0 = 0u, kp1 = 0u;
  for (int i = 0; i < NBOX; ++i) {
    int w = i >> 5, b = i & 31;
    unsigned sw = (w < 32) ? __shfl(sup0, w, 32) : __shfl(sup1, w - 32, 32);
    unsigned vw = (w < 32) ? __shfl(v0, w, 32) : __shfl(v1, w - 32, 32);
    bool keep_i = ((vw >> b) & 1u) && !((sw >> b) & 1u);
    if (keep_i) {                    // wave-uniform branch
      const unsigned* row = M + ((size_t)c * NBOX + i) * WRDS;
      sup0 |= row[lane];
      if (lane < 16) sup1 |= row[32 + lane];
      if (w < 32) { if (lane == w) kp0 |= (1u << b); }
      else        { if (lane == w - 32) kp1 |= (1u << b); }
    }
  }
  keepw[c * WRDS + lane] = kp0;
  if (lane < 16) keepw[c * WRDS + 32 + lane] = kp1;
}

// ---------------- Kernel 5: compact kept score keys (positive float bits are order-preserving) ----------------
__global__ void k_compact(const float* __restrict__ sS, const unsigned* __restrict__ keepw,
                          unsigned* hdr, unsigned* __restrict__ keys) {
  int idx = blockIdx.x * 256 + threadIdx.x;
  if (idx >= NTOT) return;
  int c = idx / NBOX, r = idx % NBOX;
  if ((keepw[c * WRDS + (r >> 5)] >> (r & 31)) & 1u) {
    unsigned pos = atomicAdd(&hdr[1], 1u);
    keys[pos] = __float_as_uint(sS[idx]);
  }
}

// ---------------- Kernel 6: exact kth-largest via bitwise descent; WMMA IU8 predicate count ----------------
__global__ __launch_bounds__(256) void k_select(const unsigned* __restrict__ keys, unsigned* hdr) {
  __shared__ int partial[8];
  __shared__ int totS;
  int tid = threadIdx.x, lane = tid & 31, wv = tid >> 5;
  int n = (int)hdr[1];
  int nch = (n + 31) >> 5;            // 32-key chunks
  int nrounds = (nch + 255) >> 8;     // chunks per 256-thread round
  const v8i Bones = {0x01010101, 0x01010101, 0x01010101, 0x01010101,
                     0x01010101, 0x01010101, 0x01010101, 0x01010101};
  const v8i Cz = {0, 0, 0, 0, 0, 0, 0, 0};
  unsigned cur = 0u;
  for (int bit = 31; bit >= 0; --bit) {
    unsigned cand = cur | (1u << bit);          // cand >= 1, so zero-padded keys never count
    int wave_tot = 0;
    for (int r = 0; r < nrounds; ++r) {
      int base = (r * 256 + tid) << 5;          // 128B-aligned; padding guarantees in-bounds
      const uint4* kp = (const uint4*)(keys + base);
      unsigned aw[8];
      #pragma unroll
      for (int q = 0; q < 8; ++q) {             // 8 x b128 loads, branch-free predicate pack
        uint4 kv = kp[q];
        aw[q] = (unsigned)(kv.x >= cand)
              | ((unsigned)(kv.y >= cand) << 8)
              | ((unsigned)(kv.z >= cand) << 16)
              | ((unsigned)(kv.w >= cand) << 24);
      }
      v8i A = {(int)aw[0], (int)aw[1], (int)aw[2], (int)aw[3],
               (int)aw[4], (int)aw[5], (int)aw[6], (int)aw[7]};
      // D[m][n] = sum_k A[m][k] * 1  -> one WMMA counts 1024 predicate bytes per wave
      v8i Dm = __builtin_amdgcn_wmma_i32_16x16x64_iu8(false, A, false, Bones, Cz, false, false);
      int acc = Dm[0] + Dm[1] + Dm[2] + Dm[3] + Dm[4] + Dm[5] + Dm[6] + Dm[7];
      wave_tot += __builtin_amdgcn_readlane(acc, 0) + __builtin_amdgcn_readlane(acc, 16);
    }
    __syncthreads();
    if (lane == 0) partial[wv] = wave_tot;
    __syncthreads();
    if (tid == 0) { int s = 0; for (int q = 0; q < 8; ++q) s += partial[q]; totS = s; }
    __syncthreads();
    if (totS >= MAXDET) cur = cand;             // monotone predicate: max T with count(>=T) >= 100
  }
  if (tid == 0) hdr[0] = cur;                   // = bits of kth largest kept score
}

// ---------------- Kernel 7: final outputs (dets | cls_idx | keep) ----------------
__global__ void k_output(const float* __restrict__ sB, const float* __restrict__ sS,
                         const unsigned* __restrict__ keepw, const unsigned* __restrict__ hdr,
                         float* __restrict__ out) {
  int idx = blockIdx.x * 256 + threadIdx.x;
  if (idx >= NTOT) return;
  int c = idx / NBOX, r = idx % NBOX;
  bool kb = (keepw[c * WRDS + (r >> 5)] >> (r & 31)) & 1u;
  float s = sS[idx];
  unsigned n = hdr[1], kth = hdr[0];
  bool keep = kb && ((n > (unsigned)MAXDET) ? (__float_as_uint(s) >= kth) : true);
  const float* b = sB + (size_t)idx * 4;
  float* dr = out + (size_t)idx * 6;
  if (keep) { dr[0] = 0.0f; dr[1] = b[0]; dr[2] = b[1]; dr[3] = b[2]; dr[4] = b[3]; dr[5] = s; }
  else      { dr[0] = 0.0f; dr[1] = 0.0f; dr[2] = 0.0f; dr[3] = 0.0f; dr[4] = 0.0f; dr[5] = 0.0f; }
  out[(size_t)NTOT * 6 + idx] = (float)(c + 1);          // cls_idx
  out[(size_t)NTOT * 7 + idx] = keep ? 1.0f : 0.0f;      // keep
}

extern "C" void kernel_launch(void* const* d_in, const int* in_sizes, int n_in,
                              void* d_out, int out_size, void* d_ws, size_t ws_size,
                              hipStream_t stream) {
  (void)in_sizes; (void)n_in; (void)out_size; (void)ws_size;
  const float* rois     = (const float*)d_in[0];
  const float* deltas   = (const float*)d_in[1];
  const float* cls_prob = (const float*)d_in[2];
  const float* im_info  = (const float*)d_in[3];
  float* out = (float*)d_out;

  // scratch carve (floats/uints), ~28.4 MB total
  float* decB   = (float*)d_ws;                       // 120000*4
  float* scores = decB + (size_t)NTOT * 4;            // 120000
  float* sB     = scores + NTOT;                      // 120000*4
  float* sS     = sB + (size_t)NTOT * 4;              // 120000
  unsigned* M     = (unsigned*)(sS + NTOT);           // 120000*48
  unsigned* keepw = M + (size_t)NTOT * WRDS;          // 80*48
  unsigned* hdr   = keepw + (size_t)NFG * WRDS;       // [0]=kth bits, [1]=n_keep
  unsigned* keys  = hdr + 8;                          // NTOT + KEYPAD (zero-padded)

  int g = (NTOT + 255) / 256;
  int gi = (NTOT + KEYPAD + 255) / 256;
  k_init<<<gi, 256, 0, stream>>>(hdr, keys);
  k_decode<<<g, 256, 0, stream>>>(rois, deltas, cls_prob, im_info, decB, scores);
  k_rank_scatter<<<g, 256, 0, stream>>>(decB, scores, sB, sS);
  k_iou_mask<<<NFG * 12, 256, 0, stream>>>(sB, M);
  k_nms_scan<<<NFG, 32, 0, stream>>>(sS, M, keepw);
  k_compact<<<g, 256, 0, stream>>>(sS, keepw, hdr, keys);
  k_select<<<1, 256, 0, stream>>>(keys, hdr);
  k_output<<<g, 256, 0, stream>>>(sB, sS, keepw, hdr, out);
}
